// GroupedQueryAttention_43224550867453
// MI455X (gfx1250) — matrix operand order repaired and checked by reading, not compiled, not run on hardware
//
#include <hip/hip_runtime.h>
#include <hip/hip_bf16.h>

typedef __attribute__((ext_vector_type(16))) _Float16 v16h;
typedef __attribute__((ext_vector_type(8)))  _Float16 v8h;
typedef __attribute__((ext_vector_type(8)))  float    v8f;

static constexpr int Bc   = 2;
static constexpr int Tc   = 2048;
static constexpr int Dc   = 1024;
static constexpr int Hc   = 16;
static constexpr int KVHc = 4;
static constexpr int HDc  = 64;

union HV16 { v16h v; v8h h[2]; };

// ---------------------------------------------------------------------------
// WMMA fragment loaders (CDNA5 ISA 7.12.2 layouts, wave32).
// All per-lane data is contiguous 8-half runs -> lowers to b128 loads.
// ---------------------------------------------------------------------------

// A-matrix 16x32 f16 from row-major (M x K): lane m = lane&15, half = lane>>4.
// Lane data = A[m][half*8 .. +7] and A[m][16+half*8 .. +7].
__device__ __forceinline__ v16h load_a16x32(const _Float16* A, int lda) {
  const int lane = threadIdx.x & 31;
  const _Float16* p = A + (size_t)(lane & 15) * lda + (lane >> 4) * 8;
  HV16 u;
  u.h[0] = *(const v8h*)(p);
  u.h[1] = *(const v8h*)(p + 16);
  return u.v;
}

// B-matrix 32x16 f16 from K-contiguous source (row-major N x K, i.e. B^T):
// lane n = lane&15, half = lane>>4; Bt[n][half*8 .. +7] and Bt[n][16+half*8 .. +7].
__device__ __forceinline__ v16h load_bt16(const _Float16* Bt, int ldb) {
  const int lane = threadIdx.x & 31;
  const _Float16* p = Bt + (size_t)(lane & 15) * ldb + (lane >> 4) * 8;
  HV16 u;
  u.h[0] = *(const v8h*)(p);
  u.h[1] = *(const v8h*)(p + 16);
  return u.v;
}

__device__ __forceinline__ v8f wmma32(v16h a, v16h b, v8f c) {
  return __builtin_amdgcn_wmma_f32_16x16x32_f16(false, a, false, b, (short)0, c,
                                                false, false);
}

// ---------------------------------------------------------------------------
// f32 -> f16 conversion (plain, and transposing for weights)
// ---------------------------------------------------------------------------
__global__ void f32_to_f16_kernel(const float* __restrict__ src,
                                  _Float16* __restrict__ dst, int n) {
  int i = blockIdx.x * blockDim.x + threadIdx.x;
  if (i < n) dst[i] = (_Float16)src[i];
}

// src: row-major K x N f32 ; dst: row-major N x K f16 (B^T)
__global__ void f32_to_f16_T_kernel(const float* __restrict__ src,
                                    _Float16* __restrict__ dst, int K, int N) {
  int i = blockIdx.x * blockDim.x + threadIdx.x;
  if (i >= K * N) return;
  int n = i % N;
  int k = i / N;
  dst[(size_t)n * K + k] = (_Float16)src[i];
}

// V: [B*T][KVH*64] -> Vt: [B][KVH][64][T]  (t-contiguous for PV B-fragments)
__global__ void transpose_v_kernel(const _Float16* __restrict__ V,
                                   _Float16* __restrict__ Vt) {
  int i = blockIdx.x * blockDim.x + threadIdx.x;   // over B*T*256
  if (i >= Bc * Tc * KVHc * HDc) return;
  int col = i & 255;
  int bt  = i >> 8;
  int b   = bt >> 11;           // /Tc
  int t   = bt & (Tc - 1);
  int kvh = col >> 6;
  int d   = col & 63;
  Vt[(((size_t)(b * KVHc + kvh)) * HDc + d) * Tc + t] = V[(size_t)bt * 256 + col];
}

// ---------------------------------------------------------------------------
// WMMA GEMM: C[MxN] = A[MxK] * B[KxN] with B supplied TRANSPOSED (Bt = NxK).
// Block = 256 threads (8 waves). Block tile 128(M) x 64(N); wave w owns rows
// m0 = by*128 + w*16 across 64 N columns. Bt k-slab (64n x 32k) staged in LDS
// K-contiguous: fill = 1 global b128 + 1 ds_store b128 per thread; consume =
// 2 ds_load b128 per fragment.
// ---------------------------------------------------------------------------
__global__ __launch_bounds__(256) void gemm_wmma_f16(
    const _Float16* __restrict__ A, const _Float16* __restrict__ Bt,
    _Float16* __restrict__ Ch, float* __restrict__ Cf, int M, int N, int K) {
  __shared__ _Float16 bs[64][40];   // 80B rows: 16B-aligned, bank-skewed
  const int w    = threadIdx.x >> 5;
  const int lane = threadIdx.x & 31;
  const int n0   = blockIdx.x * 64;
  const int m0   = blockIdx.y * 128 + w * 16;

  const int fn = threadIdx.x >> 2;        // 0..63 : n within tile
  const int fk = (threadIdx.x & 3) * 8;   // 0,8,16,24 : k chunk

  v8f acc[4] = {};

  for (int k0 = 0; k0 < K; k0 += 32) {
    // cooperative LDS fill of Bt[n0:n0+64, k0:k0+32] (K-contiguous)
    *(v8h*)&bs[fn][fk] = *(const v8h*)(Bt + (size_t)(n0 + fn) * K + k0 + fk);
    if (k0 + 32 < K)
      __builtin_prefetch(Bt + (size_t)(n0 + fn) * K + k0 + 32 + fk, 0, 1);
    __syncthreads();

    v16h a = load_a16x32(A + (size_t)m0 * K + k0, K);
#pragma unroll
    for (int c = 0; c < 4; ++c) {
      v16h b = load_bt16(&bs[c * 16][0], 40);
      acc[c] = wmma32(a, b, acc[c]);
    }
    __syncthreads();
  }

  const int half = lane >> 4;
  const int n    = lane & 15;
#pragma unroll
  for (int c = 0; c < 4; ++c) {
#pragma unroll
    for (int r = 0; r < 8; ++r) {
      int row = m0 + r + 8 * half;
      int col = n0 + c * 16 + n;
      if (Cf) Cf[(size_t)row * N + col] = acc[c][r];
      else    Ch[(size_t)row * N + col] = (_Float16)acc[c][r];
    }
  }
}

// ---------------------------------------------------------------------------
// RoPE (in place on f16 [B, T, nheads, 64])
// ---------------------------------------------------------------------------
__global__ void rope_kernel(_Float16* __restrict__ buf, int nheads, int total) {
  int idx = blockIdx.x * blockDim.x + threadIdx.x;
  if (idx >= total) return;
  int pair = idx & 31;
  int rest = idx >> 5;                 // = (b*T + t)*nheads + head
  int bt   = rest / nheads;
  int t    = bt & (Tc - 1);
  float freq = __powf(500000.0f, -(2.0f * (float)pair) / 64.0f);
  float ang  = (float)t * freq;
  float c = __cosf(ang), s = __sinf(ang);
  size_t off = (size_t)rest * 64 + (size_t)pair * 2;
  float x0 = (float)buf[off], x1 = (float)buf[off + 1];
  buf[off]     = (_Float16)(x0 * c - x1 * s);
  buf[off + 1] = (_Float16)(x1 * c + x0 * s);
}

// ---------------------------------------------------------------------------
// Flash attention (causal, GQA 4:1). Block = 8 independent waves; wave owns a
// 16-row Q tile of one (b, h). K-blocks of 32 key positions: 4 WMMAs for S
// (two 16x16 tiles x two d-steps), online softmax in C-fragment layout
// (row = r + 8*(lane>>4), col = lane&15), P (16x32) bounced through LDS into
// a full A-fragment, 4 WMMAs for O += P*V using t-contiguous Vt.
// ---------------------------------------------------------------------------
__global__ __launch_bounds__(256) void flash_attn_wmma(
    const _Float16* __restrict__ Q, const _Float16* __restrict__ Kb,
    const _Float16* __restrict__ Vt, _Float16* __restrict__ O) {
  __shared__ _Float16 plds[8][16][40];   // 80B rows
  const int w    = threadIdx.x >> 5;
  const int lane = threadIdx.x & 31;
  const int half = lane >> 4;
  const int n    = lane & 15;

  const int bh  = blockIdx.x;          // 0 .. B*H-1
  const int b   = bh >> 4;
  const int h   = bh & 15;
  const int kvh = h >> 2;              // n_rep = 4
  const int q0  = (blockIdx.y * 8 + w) * 16;

  const int ldq = Hc * HDc;            // 1024
  const int ldk = KVHc * HDc;          // 256
  const _Float16* Qp  = Q  + (size_t)(b * Tc + q0) * ldq + h * HDc;
  const _Float16* Kp  = Kb + (size_t)b * Tc * ldk + kvh * HDc;
  const _Float16* Vtp = Vt + ((size_t)(b * KVHc + kvh) * HDc) * Tc;

  v16h qa0 = load_a16x32(Qp,      ldq);   // Q[:, 0:32]
  v16h qa1 = load_a16x32(Qp + 32, ldq);   // Q[:, 32:64]

  v8f acc[4] = {};
  float mi[8], li[8];
#pragma unroll
  for (int r = 0; r < 8; ++r) { mi[r] = -3.0e38f; li[r] = 0.0f; }

  const float scale = 0.125f;          // 1/sqrt(64)
  const int jmax = q0 >> 5;            // 32-wide blocks, 0..diag

  for (int j = 0; j <= jmax; ++j) {
    const _Float16* Kj0 = Kp + (size_t)(j * 32) * ldk;       // keys j*32 .. +15
    const _Float16* Kj1 = Kj0 + (size_t)16 * ldk;            // keys +16 .. +31

    v8f s0 = {}, s1 = {};
    s0 = wmma32(qa0, load_bt16(Kj0,      ldk), s0);
    s0 = wmma32(qa1, load_bt16(Kj0 + 32, ldk), s0);
    s1 = wmma32(qa0, load_bt16(Kj1,      ldk), s1);
    s1 = wmma32(qa1, load_bt16(Kj1 + 32, ldk), s1);

    float sv0[8], sv1[8], mrow[8];
#pragma unroll
    for (int r = 0; r < 8; ++r) {
      int qr  = q0 + r + 8 * half;
      int kc0 = j * 32 + n;
      float a = s0[r] * scale;
      float c = s1[r] * scale;
      sv0[r] = (kc0      <= qr) ? a : -1.0e9f;
      sv1[r] = (kc0 + 16 <= qr) ? c : -1.0e9f;
      mrow[r] = fmaxf(sv0[r], sv1[r]);
    }
#pragma unroll
    for (int r = 0; r < 8; ++r) {
      float v = mrow[r];
      v = fmaxf(v, __shfl_xor(v, 1, 32));
      v = fmaxf(v, __shfl_xor(v, 2, 32));
      v = fmaxf(v, __shfl_xor(v, 4, 32));
      v = fmaxf(v, __shfl_xor(v, 8, 32));
      mrow[r] = v;
    }
#pragma unroll
    for (int r = 0; r < 8; ++r) {
      float mn    = fmaxf(mi[r], mrow[r]);
      float alpha = __expf(mi[r] - mn);
      float p0    = __expf(sv0[r] - mn);
      float p1    = __expf(sv1[r] - mn);
      float rs    = p0 + p1;
      rs += __shfl_xor(rs, 1, 32);
      rs += __shfl_xor(rs, 2, 32);
      rs += __shfl_xor(rs, 4, 32);
      rs += __shfl_xor(rs, 8, 32);
      li[r] = li[r] * alpha + rs;
      mi[r] = mn;
#pragma unroll
      for (int c = 0; c < 4; ++c) acc[c][r] *= alpha;
      int row = r + 8 * half;
      plds[w][row][n]      = (_Float16)p0;
      plds[w][row][n + 16] = (_Float16)p1;
    }

    // P (16x32) as full A-fragment; V block (32t x 64d) from t-contiguous Vt
    v16h pa = load_a16x32(&plds[w][0][0], 40);
    const _Float16* Vj = Vtp + (size_t)(j * 32);
#pragma unroll
    for (int c = 0; c < 4; ++c) {
      v16h vb = load_bt16(Vj + (size_t)(c * 16) * Tc, Tc);
      acc[c] = wmma32(pa, vb, acc[c]);
    }
  }

#pragma unroll
  for (int c = 0; c < 4; ++c) {
#pragma unroll
    for (int r = 0; r < 8; ++r) {
      int qr  = q0 + r + 8 * half;
      int col = h * HDc + c * 16 + n;
      O[(size_t)(b * Tc + qr) * (Hc * HDc) + col] =
          (_Float16)(acc[c][r] / li[r]);
    }
  }
}

// ---------------------------------------------------------------------------
// Launch
// ---------------------------------------------------------------------------
extern "C" void kernel_launch(void* const* d_in, const int* in_sizes, int n_in,
                              void* d_out, int out_size, void* d_ws, size_t ws_size,
                              hipStream_t stream) {
  const float* x  = (const float*)d_in[0];
  // d_in[1] = mask (implicit causal, unused)
  const float* Wq = (const float*)d_in[2];
  const float* Wk = (const float*)d_in[3];
  const float* Wv = (const float*)d_in[4];
  const float* Wo = (const float*)d_in[5];

  const int M = Bc * Tc;                 // 4096
  char* ws = (char*)d_ws;
  size_t off = 0;
  _Float16* hX   = (_Float16*)(ws + off); off += (size_t)M * Dc * 2;            // 8 MiB
  _Float16* hWqT = (_Float16*)(ws + off); off += (size_t)Dc * (Hc * HDc) * 2;   // 2 MiB
  _Float16* hWkT = (_Float16*)(ws + off); off += (size_t)Dc * (KVHc * HDc) * 2;
  _Float16* hWvT = (_Float16*)(ws + off); off += (size_t)Dc * (KVHc * HDc) * 2;
  _Float16* hWoT = (_Float16*)(ws + off); off += (size_t)(Hc * HDc) * Dc * 2;   // 2 MiB
  _Float16* Qb   = (_Float16*)(ws + off); off += (size_t)M * (Hc * HDc) * 2;    // 8 MiB
  _Float16* Kbuf = (_Float16*)(ws + off); off += (size_t)M * (KVHc * HDc) * 2;
  _Float16* Vbuf = (_Float16*)(ws + off); off += (size_t)M * (KVHc * HDc) * 2;
  _Float16* Vt   = (_Float16*)(ws + off); off += (size_t)M * (KVHc * HDc) * 2;
  _Float16* AO   = (_Float16*)(ws + off); off += (size_t)M * (Hc * HDc) * 2;    // 8 MiB

  const int thr = 256;
  // 1) f32 -> f16 conversions (weights transposed to N x K)
  f32_to_f16_kernel<<<(M * Dc + thr - 1) / thr, thr, 0, stream>>>(x, hX, M * Dc);
  f32_to_f16_T_kernel<<<(Dc * Hc * HDc + thr - 1) / thr, thr, 0, stream>>>(
      Wq, hWqT, Dc, Hc * HDc);
  f32_to_f16_T_kernel<<<(Dc * KVHc * HDc + thr - 1) / thr, thr, 0, stream>>>(
      Wk, hWkT, Dc, KVHc * HDc);
  f32_to_f16_T_kernel<<<(Dc * KVHc * HDc + thr - 1) / thr, thr, 0, stream>>>(
      Wv, hWvT, Dc, KVHc * HDc);
  f32_to_f16_T_kernel<<<(Hc * HDc * Dc + thr - 1) / thr, thr, 0, stream>>>(
      Wo, hWoT, Hc * HDc, Dc);

  // 2) QKV projections (WMMA)
  gemm_wmma_f16<<<dim3((Hc * HDc) / 64, M / 128), thr, 0, stream>>>(
      hX, hWqT, Qb, nullptr, M, Hc * HDc, Dc);
  gemm_wmma_f16<<<dim3((KVHc * HDc) / 64, M / 128), thr, 0, stream>>>(
      hX, hWkT, Kbuf, nullptr, M, KVHc * HDc, Dc);
  gemm_wmma_f16<<<dim3((KVHc * HDc) / 64, M / 128), thr, 0, stream>>>(
      hX, hWvT, Vbuf, nullptr, M, KVHc * HDc, Dc);

  // 3) RoPE on Q and K; transpose V to [B][KVH][64][T]
  {
    int totQ = M * Hc * 32;
    int totK = M * KVHc * 32;
    rope_kernel<<<(totQ + thr - 1) / thr, thr, 0, stream>>>(Qb, Hc, totQ);
    rope_kernel<<<(totK + thr - 1) / thr, thr, 0, stream>>>(Kbuf, KVHc, totK);
    int totV = M * KVHc * HDc;
    transpose_v_kernel<<<(totV + thr - 1) / thr, thr, 0, stream>>>(Vbuf, Vt);
  }

  // 4) causal flash attention (WMMA)
  flash_attn_wmma<<<dim3(Bc * Hc, Tc / 128), thr, 0, stream>>>(Qb, Kbuf, Vt, AO);

  // 5) output projection -> f32 d_out
  gemm_wmma_f16<<<dim3(Dc / 64, M / 128), thr, 0, stream>>>(
      AO, hWoT, nullptr, (float*)d_out, M, Dc, Hc * HDc);
}
